// SimpleRetrieval_70849780515074
// MI455X (gfx1250) — matrix operand order repaired and checked by reading, not compiled, not run on hardware
//
#include <hip/hip_runtime.h>
#include <hip/hip_bf16.h>
#include <stdint.h>

// Problem constants (fixed by the reference harness)
#define D_DIM   512
#define N_DB    100000
#define B_Q     1024
#define TOPK    70
#define NSPLIT  32            // N chunks -> one block column each
#define CHUNK_N 3125          // 32 * 3125 = 100000 exactly
#define TILE_M  32            // queries per block (2 WMMA M-tiles)
#define TILE_N  128           // 8 waves x 16 cols
#define TILES_PER_CHUNK 25    // 25*128 = 3200 >= 3125 (guarded)
#define KFRAGS  16            // 512 / 32
#define INV_TEMP 20.0f        // 1 / 0.05

typedef __attribute__((ext_vector_type(16))) __bf16 v16bf;
typedef __attribute__((ext_vector_type(8)))  float  v8f;

union Frag16 {
  v16bf v;
  uint4 u[2];
};

__device__ __forceinline__ unsigned short f32_to_bf16(float f) {
  unsigned int u = __float_as_uint(f);
  u += 0x7FFFu + ((u >> 16) & 1u);   // round to nearest even
  return (unsigned short)(u >> 16);
}

// ---------------------------------------------------------------------------
// Kernel 1/2: row-wise L2 normalize fp32 -> bf16.  One block per row (D=512).
// ---------------------------------------------------------------------------
__global__ __launch_bounds__(256) void l2norm_bf16_kernel(
    const float* __restrict__ src, unsigned short* __restrict__ dst) {
  const int row = blockIdx.x;
  const int tid = threadIdx.x;
  __shared__ float red[256];

  const float* s = src + (size_t)row * D_DIM;
  float x0 = s[tid];
  float x1 = s[tid + 256];
  red[tid] = x0 * x0 + x1 * x1;
  __syncthreads();
  for (int st = 128; st > 0; st >>= 1) {
    if (tid < st) red[tid] += red[tid + st];
    __syncthreads();
  }
  float scale = 1.0f / fmaxf(sqrtf(red[0]), 1e-12f);
  unsigned short* d = dst + (size_t)row * D_DIM;
  d[tid]       = f32_to_bf16(x0 * scale);
  d[tid + 256] = f32_to_bf16(x1 * scale);
}

// ---------------------------------------------------------------------------
// Kernel 3: fused sims GEMM (bf16 WMMA, fp32 accum) + per-chunk top-70.
// Block = 256 threads = 8 waves.  blockIdx.x = query block (32 q),
// blockIdx.y = N chunk (3125 rows).  Wave w owns columns [w*16, w*16+16).
//
// M=32 per block: two A-fragment sets share every streamed B fragment
// (32 FLOP per L2 byte instead of 16 -> halves the dominant L2 traffic).
// sched_barrier(0) pins all 32 B b128 loads ahead of the 32-WMMA chain so
// the backend emits staggered s_wait_loadcnt, overlapping load latency
// with the two independent accumulation chains.
// ---------------------------------------------------------------------------
__global__ __launch_bounds__(256) void sims_topk_kernel(
    const unsigned short* __restrict__ qbf,   // [1024][512] bf16
    const unsigned short* __restrict__ dbf,   // [100000][512] bf16
    float* __restrict__ cand_val,             // [1024][NSPLIT][TOPK]
    int*   __restrict__ cand_idx) {
  __shared__ float s_sims[TILE_M * TILE_N];
  __shared__ float s_vals[TILE_M][TOPK];
  __shared__ int   s_idxs[TILE_M][TOPK];
  __shared__ float s_minv[TILE_M];
  __shared__ int   s_minp[TILE_M];

  const int tid  = threadIdx.x;
  const int wave = tid >> 5;
  const int lane = tid & 31;
  const int lr   = lane & 15;          // row-in-tile (A) / col-in-tile (B)
  const int hi   = lane >> 4;          // lane half selects K sub-pattern

  const int q0          = blockIdx.x * TILE_M;
  const int n_chunk0    = blockIdx.y * CHUNK_N;
  const int n_chunk_end = n_chunk0 + CHUNK_N;   // exact (32*3125 = N)

  for (int i = tid; i < TILE_M * TOPK; i += 256) {
    (&s_vals[0][0])[i] = -1e30f;
    (&s_idxs[0][0])[i] = 0;
  }
  if (tid < TILE_M) { s_minv[tid] = -1e30f; s_minp[tid] = 0; }
  __syncthreads();

  // --- A fragments: tile-invariant, load once into registers ---------------
  // A layout (16x32 bf16): lane holds row lr, K = {a_sel..a_sel+7} and
  // {a_sel+16..a_sel+23}, a_sel = 8*hi  -> two contiguous 16B loads.
  const unsigned short* qrow0 = qbf + (size_t)(q0 + lr) * D_DIM;
  const unsigned short* qrow1 = qbf + (size_t)(q0 + 16 + lr) * D_DIM;
  const int a_sel = hi * 8;
  const int b_sel = hi * 8;    // B: lane holds col lr, K = {b_sel..b_sel+7} and {b_sel+16..b_sel+23}

  Frag16 afrag0[KFRAGS], afrag1[KFRAGS];
#pragma unroll
  for (int kk = 0; kk < KFRAGS; ++kk) {
    const unsigned short* ap0 = qrow0 + kk * 32 + a_sel;
    afrag0[kk].u[0] = *(const uint4*)(ap0);
    afrag0[kk].u[1] = *(const uint4*)(ap0 + 16);
    const unsigned short* ap1 = qrow1 + kk * 32 + a_sel;
    afrag1[kk].u[0] = *(const uint4*)(ap1);
    afrag1[kk].u[1] = *(const uint4*)(ap1 + 16);
  }

  for (int t = 0; t < TILES_PER_CHUNK; ++t) {
    const int ntile0 = n_chunk0 + t * TILE_N;
    int nrow = ntile0 + wave * 16 + lr;
    int nrow_c = (nrow < N_DB) ? nrow : 0;   // clamp; guarded at selection
    const unsigned short* drow = dbf + (size_t)nrow_c * D_DIM;

    // Issue ALL B-fragment loads first; the sched_barrier below prevents
    // the scheduler from sinking them back to their uses.
    Frag16 bfrag[KFRAGS];
#pragma unroll
    for (int kk = 0; kk < KFRAGS; ++kk) {
      const unsigned short* bp = drow + kk * 32 + b_sel;
      bfrag[kk].u[0] = *(const uint4*)(bp);
      bfrag[kk].u[1] = *(const uint4*)(bp + 16);
    }

    __builtin_amdgcn_sched_barrier(0);   // loads stay above, WMMAs below

    v8f acc0 = {};
    v8f acc1 = {};
#pragma unroll
    for (int kk = 0; kk < KFRAGS; ++kk) {
      acc0 = __builtin_amdgcn_wmma_f32_16x16x32_bf16(
          false, afrag0[kk].v, false, bfrag[kk].v, (short)0, acc0, false, false);
      acc1 = __builtin_amdgcn_wmma_f32_16x16x32_bf16(
          false, afrag1[kk].v, false, bfrag[kk].v, (short)0, acc1, false, false);
    }

    __builtin_amdgcn_sched_barrier(0);   // keep LDS staging out of the chain

    // C layout: lane holds N = lr (+wave*16), M = g + 8*hi (acc1: +16)
#pragma unroll
    for (int g = 0; g < 8; ++g) {
      s_sims[(g + hi * 8) * TILE_N + wave * 16 + lr]        = acc0[g];
      s_sims[(g + hi * 8 + 16) * TILE_N + wave * 16 + lr]   = acc1[g];
    }
    __syncthreads();

    // Selection: each half-wave (leader = lane lr==0) owns queries
    // (2*wave + hi) and (2*wave + hi + 16).
    int nmax = n_chunk_end - ntile0;
    if (nmax > TILE_N) nmax = TILE_N;
#pragma unroll
    for (int pass = 0; pass < 2; ++pass) {
      const int q = wave * 2 + hi + pass * 16;
      const float* row = &s_sims[q * TILE_N];
      float minv = s_minv[q];

      // ballot pre-filter: each lane checks 8 candidates vs running min
      bool any = false;
      for (int j = 0; j < 8; ++j) {
        int c = lr * 8 + j;
        if (c < nmax && row[c] > minv) any = true;
      }
      unsigned long long bal = __ballot(any);
      unsigned int halfmask =
          (unsigned int)(hi ? (bal >> 16) : bal) & 0xFFFFu;

      if (lr == 0 && halfmask != 0u) {
        float* vals = s_vals[q];
        int*   idxs = s_idxs[q];
        int    minp = s_minp[q];
        for (int j = 0; j < nmax; ++j) {
          float v = row[j];
          if (v > minv) {
            vals[minp] = v;
            idxs[minp] = ntile0 + j;
            minv = vals[0]; minp = 0;
            for (int kk = 1; kk < TOPK; ++kk) {
              if (vals[kk] < minv) { minv = vals[kk]; minp = kk; }
            }
          }
        }
        s_minv[q] = minv;
        s_minp[q] = minp;
      }
    }
    __syncthreads();
  }

  // Emit this chunk's 70 candidates per query.
  for (int i = tid; i < TILE_M * TOPK; i += 256) {
    int q = i / TOPK, k = i % TOPK;
    size_t off = ((size_t)(q0 + q) * NSPLIT + blockIdx.y) * TOPK + k;
    cand_val[off] = s_vals[q][k];
    cand_idx[off] = s_idxs[q][k];
  }
}

// ---------------------------------------------------------------------------
// Kernel 4: exact top-70 merge over 32*70 candidates, softmax, gather es_db.
// One block (256 threads) per query.
// ---------------------------------------------------------------------------
__global__ __launch_bounds__(256) void merge_softmax_gather_kernel(
    const float* __restrict__ cand_val,
    const int*   __restrict__ cand_idx,
    const float* __restrict__ es_db,
    float* __restrict__ out) {
  const int q   = blockIdx.x;
  const int tid = threadIdx.x;
  const int NC  = NSPLIT * TOPK;   // 2240

  __shared__ float s_v[NSPLIT * TOPK];
  __shared__ int   s_i[NSPLIT * TOPK];
  __shared__ float r_v[256];
  __shared__ int   r_p[256];
  __shared__ float s_selv[TOPK];
  __shared__ int   s_seli[TOPK];
  __shared__ float s_w[TOPK];

  for (int i = tid; i < NC; i += 256) {
    s_v[i] = cand_val[(size_t)q * NC + i];
    s_i[i] = cand_idx[(size_t)q * NC + i];
  }
  __syncthreads();

  for (int r = 0; r < TOPK; ++r) {
    float best = -1e30f; int bp = 0;
    for (int i = tid; i < NC; i += 256) {
      float v = s_v[i];
      if (v > best) { best = v; bp = i; }
    }
    r_v[tid] = best; r_p[tid] = bp;
    __syncthreads();
    for (int s = 128; s > 0; s >>= 1) {
      if (tid < s && r_v[tid + s] > r_v[tid]) {
        r_v[tid] = r_v[tid + s]; r_p[tid] = r_p[tid + s];
      }
      __syncthreads();
    }
    if (tid == 0) {
      int p = r_p[0];
      s_selv[r] = r_v[0];
      s_seli[r] = s_i[p];
      s_v[p] = -1e31f;   // remove
    }
    __syncthreads();
  }

  if (tid == 0) {
    float m = s_selv[0];           // descending order -> max first
    float sum = 0.0f;
    for (int k = 0; k < TOPK; ++k) {
      float e = __expf((s_selv[k] - m) * INV_TEMP);
      s_w[k] = e; sum += e;
    }
    float inv = 1.0f / sum;
    for (int k = 0; k < TOPK; ++k) s_w[k] *= inv;
  }
  __syncthreads();

  for (int d = tid; d < D_DIM; d += 256) {
    float acc = 0.0f;
    for (int k = 0; k < TOPK; ++k) {
      acc += s_w[k] * es_db[(size_t)s_seli[k] * D_DIM + d];
    }
    out[(size_t)q * D_DIM + d] = acc;
  }
}

// ---------------------------------------------------------------------------
extern "C" void kernel_launch(void* const* d_in, const int* in_sizes, int n_in,
                              void* d_out, int out_size, void* d_ws, size_t ws_size,
                              hipStream_t stream) {
  const float* query = (const float*)d_in[0];   // [1024, 512]
  const float* en_db = (const float*)d_in[1];   // [100000, 512]
  const float* es_db = (const float*)d_in[2];   // [100000, 512]
  float* out = (float*)d_out;                   // [1024, 512]

  char* ws = (char*)d_ws;
  // ws layout (all offsets 256B-aligned):
  //   qbf      :   1,048,576 B  (1024*512 bf16)
  //   dbf      : 102,400,000 B  (100000*512 bf16)  -- fits in 192MB L2
  //   cand_val :   9,175,040 B  (1024*32*70 f32)
  //   cand_idx :   9,175,040 B
  unsigned short* qbf = (unsigned short*)(ws);
  unsigned short* dbf = (unsigned short*)(ws + (size_t)1048576);
  float* cand_val = (float*)(ws + (size_t)1048576 + 102400000u);
  int*   cand_idx = (int*)  (ws + (size_t)1048576 + 102400000u + 9175040u);

  l2norm_bf16_kernel<<<B_Q,  256, 0, stream>>>(query, qbf);
  l2norm_bf16_kernel<<<N_DB, 256, 0, stream>>>(en_db, dbf);

  dim3 grid(B_Q / TILE_M, NSPLIT);
  sims_topk_kernel<<<grid, 256, 0, stream>>>(qbf, dbf, cand_val, cand_idx);

  merge_softmax_gather_kernel<<<B_Q, 256, 0, stream>>>(cand_val, cand_idx,
                                                       es_db, out);
}
